// SimpleHeteroGNN_78408922955995
// MI455X (gfx1250) — compile-verified, matrix operand order checked by reading, not run on hardware
//
#include <hip/hip_runtime.h>
#include <hip/hip_bf16.h>

// ---------------------------------------------------------------------------
// SimpleHeteroGNN on MI455X (gfx1250, wave32, WMMA bf16)
//
// Dense GEMMs: v_wmma_f32_16x16x32_bf16, activations pre-converted to bf16
// row-major (A fragment = two b128 loads), weights pre-packed lane-contiguous
// (B fragment = two b128 loads). Dual accumulators break the WMMA->WMMA
// hazard chain. Graph aggregation: bf16 gather + f32 global atomics.
// ---------------------------------------------------------------------------

#define N_CELL 200000
#define N_WELL 2000
#define E_CC   1200000
#define E_CW   200000
#define CF     16
#define WF     8
#define H      128       // feature width
#define HD     64        // H/2 dwords per bf16 row
#define OUTC   75

typedef __attribute__((ext_vector_type(16))) __bf16 v16bf;
typedef __attribute__((ext_vector_type(8)))  float  v8f;

union FragU { unsigned u[8]; v16bf v; };

__device__ __forceinline__ unsigned f2bf(float x) {
    unsigned u = __float_as_uint(x);
    u = (u + 0x7FFFu + ((u >> 16) & 1u)) >> 16;   // RNE
    return u & 0xFFFFu;
}
__device__ __forceinline__ unsigned pack2(float lo, float hi) {
    return f2bf(lo) | (f2bf(hi) << 16);
}
__device__ __forceinline__ float bf_lo(unsigned d) { return __uint_as_float(d << 16); }
__device__ __forceinline__ float bf_hi(unsigned d) { return __uint_as_float(d & 0xFFFF0000u); }

// ---------------------------------------------------------------------------
// Weight packing: W[128, ncols] f32 row-major -> bf16 B fragments, layout
// dst[ ((tileN*4+kk)*32 + lane)*8 + v ]  (per-lane 8 contiguous dwords).
// VGPR v holds bf16 pair at K0 = kk*32 + (v&3)*2 + (v>>2)*16 + hi*8.
// ---------------------------------------------------------------------------
__global__ void pack_w_kernel(const float* __restrict__ W, int ncols, int nTileN,
                              unsigned* __restrict__ dst) {
    int idx = blockIdx.x * blockDim.x + threadIdx.x;
    int total = nTileN * 4 * 32 * 8;
    if (idx >= total) return;
    int v    = idx & 7;
    int lane = (idx >> 3) & 31;
    int kk   = (idx >> 8) & 3;
    int tn   = idx >> 10;
    int n  = (lane & 15) + tn * 16;
    int hi = lane >> 4;
    int k0 = kk * 32 + (v & 3) * 2 + ((v >> 2) << 4) + hi * 8;
    unsigned lo = 0, hb = 0;
    if (n < ncols) {
        lo = f2bf(W[k0 * ncols + n]);
        hb = f2bf(W[(k0 + 1) * ncols + n]);
    }
    dst[idx] = lo | (hb << 16);
}

// ---------------------------------------------------------------------------
// Input projection straight to bf16: Out[n][64 dwords] = bf16(X @ W + b).
// ---------------------------------------------------------------------------
__global__ void proj_bf_kernel(const float* __restrict__ X, const float* __restrict__ W,
                               const float* __restrict__ b, unsigned* __restrict__ OutBF,
                               int n, int kdim) {
    int idx = blockIdx.x * blockDim.x + threadIdx.x;
    if (idx >= n * HD) return;
    int row = idx >> 6;
    int c0  = (idx & 63) * 2;
    const float* xr = X + (size_t)row * kdim;
    float s0 = b[c0], s1 = b[c0 + 1];
#pragma unroll 4
    for (int k = 0; k < kdim; k++) {
        float xv = xr[k];
        s0 = fmaf(xv, W[k * H + c0], s0);
        s1 = fmaf(xv, W[k * H + c0 + 1], s1);
    }
    OutBF[idx] = pack2(s0, s1);
}

// ---------------------------------------------------------------------------
// Edge scatter: gather 4 bf16 feats (uint2) from src row, f32-atomic-add into
// agg[dst], plus neighbor count. One thread per (edge, 4-feat chunk).
// ---------------------------------------------------------------------------
__global__ void scatter_kernel(const unsigned* __restrict__ featBF,
                               const int* __restrict__ src_idx,
                               const int* __restrict__ dst_idx, int E,
                               float* __restrict__ agg, float* __restrict__ cnt) {
    long long t = (long long)blockIdx.x * blockDim.x + threadIdx.x;
    if (t >= (long long)E * 32) return;
    int e = (int)(t >> 5);
    int q = ((int)t & 31);
    int s = src_idx[e];
    int d = dst_idx[e];
    uint2 w = *reinterpret_cast<const uint2*>(featBF + (size_t)s * HD + q * 2);
    float* a = agg + (size_t)d * H + q * 4;
    atomicAdd(a + 0, bf_lo(w.x));
    atomicAdd(a + 1, bf_hi(w.x));
    atomicAdd(a + 2, bf_lo(w.y));
    atomicAdd(a + 3, bf_hi(w.y));
    if (q == 0) atomicAdd(cnt + d, 1.0f);
}

// ---------------------------------------------------------------------------
// Mean + f32 -> bf16: out[row] = bf16(in[row] / max(cnt[row],1)).
// ---------------------------------------------------------------------------
__global__ void cvt_scale_bf_kernel(const float* __restrict__ in,
                                    const float* __restrict__ cnt,
                                    unsigned* __restrict__ outBF, int rows) {
    int idx = blockIdx.x * blockDim.x + threadIdx.x;
    if (idx >= rows * 32) return;
    int row = idx >> 5;
    int q   = idx & 31;
    float rs = 1.0f / fmaxf(cnt[row], 1.0f);
    float4 v = *reinterpret_cast<const float4*>(in + (size_t)row * H + q * 4);
    unsigned* o = outBF + (size_t)row * HD + q * 2;
    o[0] = pack2(v.x * rs, v.y * rs);
    o[1] = pack2(v.z * rs, v.w * rs);
}

// A fragment from bf16 row-major row: two contiguous b128 chunks.
__device__ __forceinline__ v16bf load_a_frag(const unsigned* rowp, int kk, int hi) {
    FragU f;
    uint4 x = *reinterpret_cast<const uint4*>(rowp + kk * 16 + hi * 4);
    uint4 y = *reinterpret_cast<const uint4*>(rowp + kk * 16 + hi * 4 + 8);
    f.u[0] = x.x; f.u[1] = x.y; f.u[2] = x.z; f.u[3] = x.w;
    f.u[4] = y.x; f.u[5] = y.y; f.u[6] = y.z; f.u[7] = y.w;
    return f.v;
}

// B fragment: per-lane 8 contiguous dwords -> two b128 loads.
__device__ __forceinline__ v16bf load_b_frag(const unsigned* base, int lane) {
    FragU f;
    uint4 x = *reinterpret_cast<const uint4*>(base + lane * 8);
    uint4 y = *reinterpret_cast<const uint4*>(base + lane * 8 + 4);
    f.u[0] = x.x; f.u[1] = x.y; f.u[2] = x.z; f.u[3] = x.w;
    f.u[4] = y.x; f.u[5] = y.y; f.u[6] = y.z; f.u[7] = y.w;
    return f.v;
}

// ---------------------------------------------------------------------------
// Fused WMMA GEMM: D = A1bf @ B1 (+ A2bf @ B2) + bias, optional ReLU.
// HAS_A2 is a compile-time flag (no per-iteration branch); two accumulators
// double the WMMA->WMMA dependency distance (hides the bf16 hazard window).
// One wave per 16x16 output tile; tile id wave-uniform -> EXEC all ones.
// ---------------------------------------------------------------------------
template <bool HAS_A2>
__global__ void wmma_gemm_kernel(const unsigned* __restrict__ A1,
                                 const unsigned* __restrict__ A2,
                                 const unsigned* __restrict__ Bp1,
                                 const unsigned* __restrict__ Bp2,
                                 const float* __restrict__ bias,
                                 float* __restrict__ OutF,
                                 unsigned short* __restrict__ OutB,
                                 int rows, int outCols, int nTileN, int doRelu) {
    int wave = (blockIdx.x * blockDim.x + threadIdx.x) >> 5;
    int lane = threadIdx.x & 31;
    int nTiles = (rows >> 4) * nTileN;
    if (wave >= nTiles) return;
    int tileM = wave / nTileN;
    int tileN = wave % nTileN;
    int m  = lane & 15;
    int hi = lane >> 4;
    int row = tileM * 16 + m;

    const unsigned* a1row = A1 + (size_t)row * HD;
    const unsigned* a2row = HAS_A2 ? (A2 + (size_t)row * HD) : (const unsigned*)0;

    v8f acc1 = {};
    v8f acc2 = {};
#pragma unroll
    for (int kk = 0; kk < 4; kk++) {
        v16bf a1 = load_a_frag(a1row, kk, hi);
        v16bf b1 = load_b_frag(Bp1 + (((size_t)tileN * 4 + kk) << 8), lane);
        acc1 = __builtin_amdgcn_wmma_f32_16x16x32_bf16(false, a1, false, b1,
                                                       (short)0, acc1, false, false);
        if constexpr (HAS_A2) {
            v16bf a2 = load_a_frag(a2row, kk, hi);
            v16bf b2 = load_b_frag(Bp2 + (((size_t)tileN * 4 + kk) << 8), lane);
            acc2 = __builtin_amdgcn_wmma_f32_16x16x32_bf16(false, a2, false, b2,
                                                           (short)0, acc2, false, false);
        }
    }

    int col = tileN * 16 + m;
    float bv = 0.0f;
    if (bias && col < outCols) bv = bias[col];
#pragma unroll
    for (int v = 0; v < 8; v++) {
        int r = tileM * 16 + v + hi * 8;      // C/D layout: VGPR v -> M = v + 8*hi
        float val = acc1[v] + bv;
        if constexpr (HAS_A2) val += acc2[v];
        if (doRelu) val = fmaxf(val, 0.0f);
        if (OutF && col < outCols) OutF[(size_t)r * outCols + col] = val;
        if (OutB && col < outCols)
            OutB[(size_t)r * H + col] = (unsigned short)f2bf(val);
    }
}

extern "C" void kernel_launch(void* const* d_in, const int* in_sizes, int n_in,
                              void* d_out, int out_size, void* d_ws, size_t ws_size,
                              hipStream_t stream) {
    (void)in_sizes; (void)n_in; (void)out_size; (void)ws_size;

    const float* cell_x  = (const float*)d_in[0];
    const float* well_x  = (const float*)d_in[1];
    const int*   e_cc    = (const int*)d_in[2];    // [2, E_CC]
    const int*   e_w_src = (const int*)d_in[3];
    const int*   e_w_dst = (const int*)d_in[4];
    const float* W_cell  = (const float*)d_in[5];
    const float* b_cell  = (const float*)d_in[6];
    const float* W_well  = (const float*)d_in[7];
    const float* b_well  = (const float*)d_in[8];
    const float* Wl_cc   = (const float*)d_in[9];
    const float* bl_cc   = (const float*)d_in[10];
    const float* Wr_cc   = (const float*)d_in[11];
    const float* Wl_cw   = (const float*)d_in[12];
    const float* bl_cw   = (const float*)d_in[13];
    const float* Wr_cw   = (const float*)d_in[14];
    const float* W_m1    = (const float*)d_in[15];
    const float* b_m1    = (const float*)d_in[16];
    const float* W_m2    = (const float*)d_in[17];
    const float* b_m2    = (const float*)d_in[18];
    float* out = (float*)d_out;

    // ---- workspace carve-up ------------------------------------------------
    char* ws = (char*)d_ws;
    size_t off = 0;
    auto carve = [&](size_t bytes) -> char* {
        char* p = ws + off;
        off += (bytes + 255) & ~(size_t)255;
        return p;
    };
    unsigned* cell_h_bf  = (unsigned*)carve((size_t)N_CELL * HD * 4);  // bf16 acts
    unsigned* cell_h2_bf = (unsigned*)carve((size_t)N_CELL * HD * 4);
    unsigned* aggc_bf    = (unsigned*)carve((size_t)N_CELL * HD * 4);
    float*    agg_c      = (float*)carve((size_t)N_CELL * H * 4);      // f32 accum
    float*    cnt_c      = (float*)carve((size_t)N_CELL * 4);
    float*    agg_w      = (float*)carve((size_t)N_WELL * H * 4);
    float*    cnt_w      = (float*)carve((size_t)N_WELL * 4);
    unsigned* well_h_bf  = (unsigned*)carve((size_t)N_WELL * HD * 4);
    unsigned* aggw_bf    = (unsigned*)carve((size_t)N_WELL * HD * 4);
    unsigned* well_h2_bf = (unsigned*)carve((size_t)N_WELL * HD * 4);
    unsigned* h_bf       = (unsigned*)carve((size_t)N_WELL * HD * 4);
    unsigned* p_Wl_cc = (unsigned*)carve(8 * 4 * 32 * 8 * 4);
    unsigned* p_Wr_cc = (unsigned*)carve(8 * 4 * 32 * 8 * 4);
    unsigned* p_Wl_cw = (unsigned*)carve(8 * 4 * 32 * 8 * 4);
    unsigned* p_Wr_cw = (unsigned*)carve(8 * 4 * 32 * 8 * 4);
    unsigned* p_Wm1   = (unsigned*)carve(8 * 4 * 32 * 8 * 4);
    unsigned* p_Wm2   = (unsigned*)carve(5 * 4 * 32 * 8 * 4);

    // ---- zero accumulators -------------------------------------------------
    hipMemsetAsync(agg_c, 0, (size_t)N_CELL * H * 4, stream);
    hipMemsetAsync(cnt_c, 0, (size_t)N_CELL * 4, stream);
    hipMemsetAsync(agg_w, 0, (size_t)N_WELL * H * 4, stream);
    hipMemsetAsync(cnt_w, 0, (size_t)N_WELL * 4, stream);

    // ---- pack weights ------------------------------------------------------
    const int pt = 8 * 4 * 32 * 8;
    pack_w_kernel<<<(pt + 255) / 256, 256, 0, stream>>>(Wl_cc, H, 8, p_Wl_cc);
    pack_w_kernel<<<(pt + 255) / 256, 256, 0, stream>>>(Wr_cc, H, 8, p_Wr_cc);
    pack_w_kernel<<<(pt + 255) / 256, 256, 0, stream>>>(Wl_cw, H, 8, p_Wl_cw);
    pack_w_kernel<<<(pt + 255) / 256, 256, 0, stream>>>(Wr_cw, H, 8, p_Wr_cw);
    pack_w_kernel<<<(pt + 255) / 256, 256, 0, stream>>>(W_m1, H, 8, p_Wm1);
    const int pt2 = 5 * 4 * 32 * 8;
    pack_w_kernel<<<(pt2 + 255) / 256, 256, 0, stream>>>(W_m2, OUTC, 5, p_Wm2);

    // ---- input projections (write bf16) ------------------------------------
    proj_bf_kernel<<<((size_t)N_CELL * HD + 255) / 256, 256, 0, stream>>>(
        cell_x, W_cell, b_cell, cell_h_bf, N_CELL, CF);
    proj_bf_kernel<<<((size_t)N_WELL * HD + 255) / 256, 256, 0, stream>>>(
        well_x, W_well, b_well, well_h_bf, N_WELL, WF);

    // ---- cell->cell aggregation -------------------------------------------
    {
        long long thr = (long long)E_CC * 32;
        scatter_kernel<<<(unsigned)((thr + 255) / 256), 256, 0, stream>>>(
            cell_h_bf, e_cc, e_cc + E_CC, E_CC, agg_c, cnt_c);
    }
    cvt_scale_bf_kernel<<<((size_t)N_CELL * 32 + 255) / 256, 256, 0, stream>>>(
        agg_c, cnt_c, aggc_bf, N_CELL);

    // ---- cell SAGE: cell_h2 = mean_agg @ Wl_cc + bl_cc + cell_h @ Wr_cc ----
    {
        int nTiles = (N_CELL / 16) * 8;   // 100000 waves
        wmma_gemm_kernel<true><<<(nTiles * 32 + 255) / 256, 256, 0, stream>>>(
            aggc_bf, cell_h_bf, p_Wl_cc, p_Wr_cc, bl_cc,
            (float*)0, (unsigned short*)cell_h2_bf, N_CELL, H, 8, 0);
    }

    // ---- cell->well aggregation -------------------------------------------
    {
        long long thr = (long long)E_CW * 32;
        scatter_kernel<<<(unsigned)((thr + 255) / 256), 256, 0, stream>>>(
            cell_h2_bf, e_w_src, e_w_dst, E_CW, agg_w, cnt_w);
    }
    cvt_scale_bf_kernel<<<((size_t)N_WELL * 32 + 255) / 256, 256, 0, stream>>>(
        agg_w, cnt_w, aggw_bf, N_WELL);

    // ---- well SAGE ---------------------------------------------------------
    {
        int nTiles = (N_WELL / 16) * 8;   // 1000 waves
        wmma_gemm_kernel<true><<<(nTiles * 32 + 255) / 256, 256, 0, stream>>>(
            aggw_bf, well_h_bf, p_Wl_cw, p_Wr_cw, bl_cw,
            (float*)0, (unsigned short*)well_h2_bf, N_WELL, H, 8, 0);
    }

    // ---- MLP head ----------------------------------------------------------
    {
        int nTiles = (N_WELL / 16) * 8;
        wmma_gemm_kernel<false><<<(nTiles * 32 + 255) / 256, 256, 0, stream>>>(
            well_h2_bf, (const unsigned*)0, p_Wm1, (const unsigned*)0, b_m1,
            (float*)0, (unsigned short*)h_bf, N_WELL, H, 8, 1);
    }
    {
        int nTiles = (N_WELL / 16) * 5;   // masked to 75 cols
        wmma_gemm_kernel<false><<<(nTiles * 32 + 255) / 256, 256, 0, stream>>>(
            h_bf, (const unsigned*)0, p_Wm2, (const unsigned*)0, b_m2,
            out, (unsigned short*)0, N_WELL, OUTC, 5, 0);
    }
}
